// simple_LSTM_79474074845609
// MI455X (gfx1250) — matrix operand order repaired and checked by reading, not compiled, not run on hardware
//
#include <hip/hip_runtime.h>
#include <hip/hip_bf16.h>

typedef __attribute__((ext_vector_type(16))) __bf16       v16bf;
typedef __attribute__((ext_vector_type(8)))  float        v8f;
typedef __attribute__((ext_vector_type(4)))  unsigned int v4u;

#define T_STEPS 512
#define BATCH   64
#define EDIM    512
#define HDIM    512
#define NWG     32          // workgroups in persistent recurrent kernel
#define CPW     16          // h-columns per WG (NWG * CPW == HDIM)
#define THREADS 256

// workspace layout (byte offsets)
#define OFF_ARRIVE 0u                   // u32[T_STEPS]
#define OFF_BIAS   (16u * 1024u)        // f32[4H]
#define OFF_H      (32u * 1024u)        // bf16, 2 parities x (64*512) = 128KB
#define OFF_W      (256u * 1024u)       // bf16, 128 ntiles x 32 ktiles x 32 lanes x 16 = 4MB
#define OFF_X      (8u * 1024u * 1024u) // bf16, T x (64*512) swizzled = 32MB

__device__ __forceinline__ unsigned short f2bf(float f) {
    unsigned u = __builtin_bit_cast(unsigned, f);
    return (unsigned short)((u + 0x7FFFu + ((u >> 16) & 1u)) >> 16);  // RNE
}

// Index (in bf16 elements) of (m,k) inside one swizzled [64 x 512] A-operand block.
// 16-bit A layout (contiguous k order): lane<16 holds K 0..15, lane>=16 holds K 16..31,
// stored so each lane's 16 values are contiguous (two 16B chunks).
__device__ __forceinline__ int a_swz(int m, int k) {
    int mt = m >> 4, lo = m & 15, kt = k >> 5, ko = k & 31;
    int lane, off;
    lane = lo + (ko >> 4) * 16;
    off  = ko & 15;
    return ((mt * 16 + kt) * 32 + lane) * 16 + off;
}

__device__ __forceinline__ float fast_sigmoid(float x) {
    return 1.0f / (1.0f + __expf(-x));
}
__device__ __forceinline__ float fast_tanh(float x) {
    float e2 = __expf(2.0f * x);
    return (e2 - 1.0f) / (e2 + 1.0f);
}

// ---------------- prep kernels ----------------

__global__ void k_init(const float* __restrict__ b_ih, const float* __restrict__ b_hh,
                       float* __restrict__ bias, unsigned* __restrict__ arrive,
                       unsigned* __restrict__ h0) {
    int i = blockIdx.x * blockDim.x + threadIdx.x;
    if (i < T_STEPS) arrive[i] = 0u;                       // graph-replay safe reset
    if (i < 4 * HDIM) bias[i] = b_ih[i] + b_hh[i];
    for (int j = i; j < (BATCH * HDIM) / 2; j += gridDim.x * blockDim.x)
        h0[j] = 0u;                                        // zero parity-0 h (bf16 pairs)
}

// W = [W_ih | W_hh] (rows g: PyTorch gate order i,f,g,o). Store bf16 B-operand layout:
// [gnt 0..127][kt 0..31][lane 0..31][16], lanes 0-15 <-> K 0..15 of tile, 16-31 <-> K 16..31.
__global__ void k_swz_w(const float* __restrict__ W_ih, const float* __restrict__ W_hh,
                        unsigned short* __restrict__ wswz) {
    int tid = blockIdx.x * blockDim.x + threadIdx.x;       // (4H) * (1024/16)
    if (tid >= 4 * HDIM * 64) return;
    int kb = tid & 63;
    int g  = tid >> 6;
    int k0 = kb * 16;
    const float* src = (k0 < EDIM) ? (W_ih + g * EDIM + k0)
                                   : (W_hh + g * HDIM + (k0 - EDIM));
    int gnt  = g >> 4;
    int kt   = k0 >> 5;
    int half = (k0 >> 4) & 1;
    int lane = half * 16 + (g & 15);
    unsigned short* dst = wswz + ((gnt * 32 + kt) * 32 + lane) * 16;
#pragma unroll
    for (int j = 0; j < 16; ++j) dst[j] = f2bf(src[j]);
}

// seq [T,B,E] fp32 -> bf16 A-operand swizzle per timestep; 8 consecutive e map to one 16B chunk.
__global__ void k_swz_x(const float* __restrict__ seq, unsigned short* __restrict__ xswz) {
    int tid = blockIdx.x * blockDim.x + threadIdx.x;       // T*B*E/8
    if (tid >= T_STEPS * BATCH * (EDIM / 8)) return;
    int e8 = tid & 63;
    int r  = (tid >> 6) & 63;
    int t  = tid >> 12;
    int e0 = e8 * 8;
    const float* src = seq + ((t * BATCH) + r) * EDIM + e0;
    __align__(16) unsigned short tmp[8];
#pragma unroll
    for (int j = 0; j < 8; ++j) tmp[j] = f2bf(src[j]);
    int idx = t * (BATCH * EDIM) + a_swz(r, e0);           // e0 % 8 == 0 -> contiguous chunk
    *(v4u*)(xswz + idx) = *(const v4u*)tmp;
}

// ---------------- persistent recurrent kernel ----------------

__global__ __launch_bounds__(THREADS, 1) void k_lstm(
    const unsigned short* __restrict__ xswz,
    const unsigned short* __restrict__ wswz,
    unsigned short* __restrict__ hswz,      // 2 parities of [64x512] swizzled bf16
    const float* __restrict__ bias,
    unsigned* __restrict__ arrive,
    float* __restrict__ out) {

    __shared__ __align__(16) unsigned short b_lds[4 * 32 * 32 * 16]; // 128KB: W^T slice, resident all steps
    __shared__ float gate_lds[64 * 64];                              // 16KB: pre-activation gates
    __shared__ float c_lds[64 * CPW];                                // 4KB : cell state

    const int wg     = blockIdx.x;        // 0..31
    const int tid    = threadIdx.x;
    const int lane   = tid & 31;
    const int wave   = tid >> 5;          // 8 waves
    const int m_tile = wave & 3;          // batch-row tile
    const int n0     = (wave >> 2) * 2;   // gate-group tiles {n0, n0+1}
    const int c0     = wg * CPW;

    // One-time: pull this WG's 4x32KB weight slices into LDS (gate group q -> gnt = q*32+wg).
    for (int i = tid; i < 4 * 2048; i += THREADS) {
        int q = i >> 11, j = i & 2047;
        ((v4u*)b_lds)[q * 2048 + j] = ((const v4u*)wswz)[(q * 32 + wg) * 2048 + j];
    }
    for (int i = tid; i < 64 * CPW; i += THREADS) c_lds[i] = 0.0f;
    __syncthreads();

    // Per-lane invariant base offsets (elements). Per-k delta = 512 elements = 1KB;
    // with the k-loop fully unrolled these become constant instruction offsets.
    const int a_base = (m_tile * 16 * 32 + lane) * 16;               // + kk*512
    const unsigned short* bbase = b_lds + (n0 * 32 * 32 + lane) * 16; // + kk*512; +16384 for n0+1

    for (int t = 0; t < T_STEPS; ++t) {
        const unsigned short* hcur = hswz + (t & 1) * (BATCH * HDIM);
        unsigned short*       hnxt = hswz + ((t + 1) & 1) * (BATCH * HDIM);
        const unsigned short* xcur = xswz + t * (BATCH * EDIM);
        const unsigned short* ax   = xcur + a_base;
        const unsigned short* ah   = hcur + a_base;

        // gates[64, 64cols] = [x_t | h] (64x1024) @ W_slice^T, K=1024 as 32 bf16 WMMA k-steps.
        // Fully unrolled: straight-line loads with immediate offsets -> deep load/WMMA overlap.
        v8f acc0 = {}, acc1 = {};
#pragma unroll
        for (int kk = 0; kk < 32; ++kk) {
            const unsigned short* ap = (kk < 16) ? (ax + kk * 512) : (ah + (kk - 16) * 512);
            union { v4u q[2]; v16bf v; } ua, ub0, ub1;
            ua.q[0] = ((const v4u*)ap)[0];
            ua.q[1] = ((const v4u*)ap)[1];
            const unsigned short* bp0 = bbase + kk * 512;
            const unsigned short* bp1 = bp0 + 32 * 32 * 16;
            ub0.q[0] = ((const v4u*)bp0)[0]; ub0.q[1] = ((const v4u*)bp0)[1];
            ub1.q[0] = ((const v4u*)bp1)[0]; ub1.q[1] = ((const v4u*)bp1)[1];
            acc0 = __builtin_amdgcn_wmma_f32_16x16x32_bf16(false, ua.v, false, ub0.v,
                                                           (short)0, acc0, false, false);
            acc1 = __builtin_amdgcn_wmma_f32_16x16x32_bf16(false, ua.v, false, ub1.v,
                                                           (short)0, acc1, false, false);
        }

        // Prefetch next step's x A-operand chunks while we head into the barrier.
        if (t + 1 < T_STEPS) {
            const unsigned short* axn = xcur + (BATCH * EDIM) + a_base;
#pragma unroll
            for (int pk = 0; pk < 16; pk += 4)
                __builtin_prefetch(axn + pk * 512, 0, 0);
        }

        // C/D layout: lane<16 -> rows r, lane>=16 -> rows r+8; col = lane&15
        {
            int col0  = (n0 + 0) * 16 + (lane & 15);
            int col1  = (n0 + 1) * 16 + (lane & 15);
            int rbase = m_tile * 16 + ((lane < 16) ? 0 : 8);
#pragma unroll
            for (int r = 0; r < 8; ++r) {
                gate_lds[(rbase + r) * 64 + col0] = acc0[r];
                gate_lds[(rbase + r) * 64 + col1] = acc1[r];
            }
        }
        __syncthreads();

        // LSTM cell update for this WG's 64 rows x 16 h-columns
        for (int idx = tid; idx < 64 * CPW; idx += THREADS) {
            int row = idx >> 4, hc = idx & 15;
            float gi = gate_lds[row * 64 + 0 * 16 + hc] + bias[0 * HDIM + c0 + hc];
            float gf = gate_lds[row * 64 + 1 * 16 + hc] + bias[1 * HDIM + c0 + hc];
            float gg = gate_lds[row * 64 + 2 * 16 + hc] + bias[2 * HDIM + c0 + hc];
            float go = gate_lds[row * 64 + 3 * 16 + hc] + bias[3 * HDIM + c0 + hc];
            float i_ = fast_sigmoid(gi);
            float f_ = fast_sigmoid(gf);
            float g_ = fast_tanh(gg);
            float o_ = fast_sigmoid(go);
            float cn = f_ * c_lds[idx] + i_ * g_;
            float hn = o_ * fast_tanh(cn);
            c_lds[idx] = cn;
            hnxt[a_swz(row, c0 + hc)] = f2bf(hn);          // next step's A operand, pre-swizzled
            if (t == T_STEPS - 1) out[row * HDIM + c0 + hc] = hn;
        }

        // device-wide step barrier (per-step counters, zeroed by k_init each launch)
        __builtin_amdgcn_fence(__ATOMIC_RELEASE, "agent");
        __syncthreads();
        if (tid == 0) {
            __hip_atomic_fetch_add(&arrive[t], 1u, __ATOMIC_RELEASE, __HIP_MEMORY_SCOPE_AGENT);
            while (__hip_atomic_load(&arrive[t], __ATOMIC_ACQUIRE, __HIP_MEMORY_SCOPE_AGENT)
                   < (unsigned)NWG)
                __builtin_amdgcn_s_sleep(1);
        }
        __syncthreads();
        __builtin_amdgcn_fence(__ATOMIC_ACQUIRE, "agent");
    }
}

extern "C" void kernel_launch(void* const* d_in, const int* in_sizes, int n_in,
                              void* d_out, int out_size, void* d_ws, size_t ws_size,
                              hipStream_t stream) {
    const float* seq  = (const float*)d_in[0];
    const float* W_ih = (const float*)d_in[1];
    const float* W_hh = (const float*)d_in[2];
    const float* b_ih = (const float*)d_in[3];
    const float* b_hh = (const float*)d_in[4];
    float* out = (float*)d_out;
    char* ws = (char*)d_ws;

    unsigned*       arrive = (unsigned*)(ws + OFF_ARRIVE);
    float*          bias   = (float*)(ws + OFF_BIAS);
    unsigned short* hswz   = (unsigned short*)(ws + OFF_H);
    unsigned short* wswz   = (unsigned short*)(ws + OFF_W);
    unsigned short* xswz   = (unsigned short*)(ws + OFF_X);

    k_init<<<64, 256, 0, stream>>>(b_ih, b_hh, bias, arrive, (unsigned*)hswz);
    k_swz_w<<<(4 * HDIM * 64 + 255) / 256, 256, 0, stream>>>(W_ih, W_hh, wswz);
    k_swz_x<<<(T_STEPS * BATCH * (EDIM / 8) + 255) / 256, 256, 0, stream>>>(seq, xswz);
    k_lstm<<<NWG, THREADS, 0, stream>>>(xswz, wswz, hswz, bias, arrive, out);
}
